// AdaptiveRegionGenerator_77395310674105
// MI455X (gfx1250) — matrix operand order repaired and checked by reading, not compiled, not run on hardware
//
#include <hip/hip_runtime.h>
#include <cstdint>
#include <cstddef>

// ---------------- problem constants ----------------
#define NPATCH 16384
#define DMODEL 768
#define DFF    2048
#define GRIDK  8
#define RGN    64          // 8x8 anchor regions
#define SMAX   384         // padded sequence (1 CLS + up to 383 patches)
#define VMAX   64
#define NHEAD  8
#define HDIM   96
#define ROWS   (VMAX * SMAX)   // 24576 token rows
#define LNEPS  1e-5f
#define QKVLD  (3 * DMODEL)    // 2304

typedef __bf16 bf16;
typedef __bf16 v16bf __attribute__((ext_vector_type(16)));
typedef float  v8f   __attribute__((ext_vector_type(8)));
typedef unsigned int u32x4 __attribute__((ext_vector_type(4)));

// ---------------- workspace layout ----------------
struct Ctrl {
    float partMin[64][2];
    float partMax[64][2];
    float tmin[2], span[2];
    float P[RGN][2];          // anchor centers
    float cnt0[RGN];          // grid-init counts (scatter mean)
    unsigned maxbits[RGN];    // float-bit atomicMax of cdist column max
    int regCount[RGN];        // assigned counts per region
    int region2slot[RGN];
    int slotCount[VMAX];      // clamped valid-key count per slot
    int cursor[VMAX];
    int V;
};

static constexpr size_t mb_align(size_t x) {
    return (x + (((size_t)1 << 20) - 1)) & ~(((size_t)1 << 20) - 1);
}
static constexpr size_t OFF_ASSIGN = (size_t)8 << 10;
static constexpr size_t OFF_SLOTPT = (size_t)80 << 10;
static constexpr size_t OFF_G      = (size_t)160 << 10;          // 64x768 f32
static constexpr size_t OFF_X      = (size_t)1 << 20;            // tokens f32
static constexpr size_t OFF_H      = mb_align(OFF_X + (size_t)ROWS * DMODEL * 4);
static constexpr size_t OFF_O      = mb_align(OFF_H + (size_t)ROWS * DMODEL * 4);
static constexpr size_t OFF_QKV    = mb_align(OFF_O + (size_t)ROWS * DMODEL * 4); // bf16 qkv / later f32 TMP
static constexpr size_t OFF_ABF    = mb_align(OFF_QKV + (size_t)ROWS * QKVLD * 2);
static constexpr size_t OFF_FF1    = mb_align(OFF_ABF + (size_t)ROWS * DMODEL * 2);
static constexpr size_t OFF_WBF    = mb_align(OFF_FF1 + (size_t)ROWS * DFF * 2);
static constexpr size_t WOFF_QKV = 0;
static constexpr size_t WOFF_OUT = WOFF_QKV + (size_t)QKVLD * DMODEL;
static constexpr size_t WOFF_L1  = WOFF_OUT + (size_t)DMODEL * DMODEL;
static constexpr size_t WOFF_L2  = WOFF_L1 + (size_t)DFF * DMODEL;

// ---------------- CDNA5 async copy helpers ----------------
// GV-mode async global->LDS copy of 16 bytes per enabled lane (tracked by ASYNCcnt).
__device__ __forceinline__ void async_g2l_b128(unsigned lds_off, const void* gaddr) {
    asm volatile("global_load_async_to_lds_b128 %0, %1, off"
                 :
                 : "v"(lds_off), "v"((unsigned long long)(size_t)gaddr)
                 : "memory");
}
__device__ __forceinline__ void wait_async0() {
    asm volatile("s_wait_asynccnt 0x0" ::: "memory");
}
__device__ __forceinline__ unsigned lds_byte_off(const void* p) {
    // generic LDS address: aperture in [63:32], LDS byte offset in [31:0]
    return (unsigned)(size_t)p;
}

// ---------------- WMMA helpers ----------------
__device__ __forceinline__ v8f wmma_bf(v16bf a, v16bf b, v8f c) {
    return __builtin_amdgcn_wmma_f32_16x16x32_bf16(false, a, false, b, (short)0, c,
                                                   false, false);
}

// A-fragment, row-major MxK source (works for global or LDS pointers).
// lane L holds row m0+(L&15); element e -> k = k0 + (L>>4)*16 + e.
__device__ __forceinline__ v16bf frag_a_rowmajor(const bf16* base, int ld, int m0,
                                                 int k0, unsigned lane) {
    const bf16* p = base + (size_t)(m0 + (int)(lane & 15)) * ld + k0 + (int)((lane >> 4) << 4);
    union { u32x4 q[2]; v16bf v; } r;
    r.q[0] = *(const u32x4*)(p);
    r.q[1] = *(const u32x4*)(p + 8);
    return r.v;
}

// B-fragment where B[k][n] = W[n][k] (weight (N,K) row-major): contiguous in k.
// lane L holds col n0+(L&15); element e -> k = k0 + (L>>4)*16 + e.
__device__ __forceinline__ v16bf frag_b_nk(const bf16* base, int ld, int n0, int k0,
                                           unsigned lane) {
    const bf16* p = base + (size_t)(n0 + (int)(lane & 15)) * ld + k0 + (int)((lane >> 4) << 4);
    union { u32x4 q[2]; v16bf v; } r;
    r.q[0] = *(const u32x4*)(p);
    r.q[1] = *(const u32x4*)(p + 8);
    return r.v;
}

// B-fragment from row-major KxN source (strided in k).
__device__ __forceinline__ v16bf frag_b_kn(const bf16* base, int ld, int k0, int n0,
                                           unsigned lane) {
    const bf16* p = base + (size_t)(k0 + (int)((lane >> 4) << 4)) * ld + n0 + (int)(lane & 15);
    v16bf r;
#pragma unroll
    for (int e = 0; e < 16; ++e) r[e] = p[(size_t)e * ld];
    return r;
}

// ---------------- utility kernels ----------------
__global__ void k_zero_u32(unsigned* __restrict__ p, long n) {
    long i = (long)blockIdx.x * blockDim.x + threadIdx.x;
    long st = (long)gridDim.x * blockDim.x;
    for (; i < n; i += st) p[i] = 0u;
}

__global__ void k_f32_to_bf16(const float* __restrict__ in, bf16* __restrict__ out, long n) {
    long i = (long)blockIdx.x * blockDim.x + threadIdx.x;
    long st = (long)gridDim.x * blockDim.x;
    for (; i < n; i += st) out[i] = (bf16)in[i];
}

// ---------------- routing ----------------
__global__ void k_minmax(const float* __restrict__ T, Ctrl* __restrict__ ctrl) {
    __shared__ float sx[2][256], sy[2][256];
    int t = threadIdx.x;
    int i = blockIdx.x * 256 + t;           // 64 blocks x 256 == NPATCH
    float x = T[2 * i], y = T[2 * i + 1];
    sx[0][t] = x; sx[1][t] = x; sy[0][t] = y; sy[1][t] = y;
    __syncthreads();
    for (int off = 128; off > 0; off >>= 1) {
        if (t < off) {
            sx[0][t] = fminf(sx[0][t], sx[0][t + off]);
            sx[1][t] = fmaxf(sx[1][t], sx[1][t + off]);
            sy[0][t] = fminf(sy[0][t], sy[0][t + off]);
            sy[1][t] = fmaxf(sy[1][t], sy[1][t + off]);
        }
        __syncthreads();
    }
    if (t == 0) {
        ctrl->partMin[blockIdx.x][0] = sx[0][0];
        ctrl->partMax[blockIdx.x][0] = sx[1][0];
        ctrl->partMin[blockIdx.x][1] = sy[0][0];
        ctrl->partMax[blockIdx.x][1] = sy[1][0];
    }
}

__global__ void k_finalize(Ctrl* __restrict__ ctrl) {
    if (threadIdx.x == 0) {
        float mn0 = ctrl->partMin[0][0], mx0 = ctrl->partMax[0][0];
        float mn1 = ctrl->partMin[0][1], mx1 = ctrl->partMax[0][1];
        for (int b = 1; b < 64; ++b) {
            mn0 = fminf(mn0, ctrl->partMin[b][0]); mx0 = fmaxf(mx0, ctrl->partMax[b][0]);
            mn1 = fminf(mn1, ctrl->partMin[b][1]); mx1 = fmaxf(mx1, ctrl->partMax[b][1]);
        }
        ctrl->tmin[0] = mn0; ctrl->tmin[1] = mn1;
        ctrl->span[0] = fmaxf(mx0 - mn0, 1.1920929e-7f);
        ctrl->span[1] = fmaxf(mx1 - mn1, 1.1920929e-7f);
    }
    __syncthreads();
    int r = threadIdx.x;
    if (r < RGN) {
        int i = r / GRIDK, j = r % GRIDK;
        ctrl->P[r][0] = ctrl->tmin[0] + ctrl->span[0] * ((i + 0.5f) / GRIDK);
        ctrl->P[r][1] = ctrl->tmin[1] + ctrl->span[1] * ((j + 0.5f) / GRIDK);
    }
}

__global__ void k_accumG(const float* __restrict__ F, const float* __restrict__ T,
                         Ctrl* __restrict__ ctrl, float* __restrict__ G) {
    __shared__ int sid;
    int i = blockIdx.x;
    if (threadIdx.x == 0) {
        float tx = T[2 * i], ty = T[2 * i + 1];
        int xb = 0, yb = 0;
        for (int j = 1; j < GRIDK; ++j) {
            if (ctrl->tmin[0] + ctrl->span[0] * (j / 8.0f) < tx) xb++;
            if (ctrl->tmin[1] + ctrl->span[1] * (j / 8.0f) < ty) yb++;
        }
        sid = xb * GRIDK + yb;
        atomicAdd(&ctrl->cnt0[sid], 1.0f);
    }
    __syncthreads();
    int id = sid;
    for (int d = threadIdx.x; d < DMODEL; d += 256)
        atomicAdd(&G[(size_t)id * DMODEL + d], F[(size_t)i * DMODEL + d]);
}

__global__ void k_normG(Ctrl* __restrict__ ctrl, float* __restrict__ G) {
    int r = blockIdx.x;
    float c = fmaxf(ctrl->cnt0[r], 1.0f);
    float inv = 1.0f / c;
    for (int d = threadIdx.x; d < DMODEL; d += 256) G[(size_t)r * DMODEL + d] *= inv;
}

__global__ void k_maxdist(const float* __restrict__ T, Ctrl* __restrict__ ctrl) {
    int i = blockIdx.x * 256 + threadIdx.x;
    if (i >= NPATCH) return;
    float tx = T[2 * i], ty = T[2 * i + 1];
    for (int r = 0; r < RGN; ++r) {
        float dx = tx - ctrl->P[r][0], dy = ty - ctrl->P[r][1];
        float dist = sqrtf(dx * dx + dy * dy);
        atomicMax(&ctrl->maxbits[r], __float_as_uint(dist));
    }
}

__global__ void k_assign(const float* __restrict__ F, const float* __restrict__ T,
                         Ctrl* __restrict__ ctrl, const float* __restrict__ G,
                         int* __restrict__ assigned) {
    __shared__ float mu_s[RGN];
    __shared__ int cand[3];
    __shared__ float muk[3];
    __shared__ float p_f2[64];
    __shared__ float p_dot[3][64];
    __shared__ float p_g2[3][64];
    int i = blockIdx.x;
    int t = threadIdx.x;
    float tx = T[2 * i], ty = T[2 * i + 1];
    if (t < RGN) {
        float dx = tx - ctrl->P[t][0], dy = ty - ctrl->P[t][1];
        float dist = sqrtf(dx * dx + dy * dy);
        float md = fmaxf(__uint_as_float(ctrl->maxbits[t]), 1e-12f);
        mu_s[t] = 1.0f - dist / md;
    }
    __syncthreads();
    if (t == 0) {
        float b; int bi;
        b = -1e30f; bi = 0;
        for (int r = 0; r < RGN; ++r) if (mu_s[r] > b) { b = mu_s[r]; bi = r; }
        cand[0] = bi; muk[0] = b;
        int c0 = bi;
        b = -1e30f; bi = 0;
        for (int r = 0; r < RGN; ++r) { if (r == c0) continue; if (mu_s[r] > b) { b = mu_s[r]; bi = r; } }
        cand[1] = bi; muk[1] = b;
        int c1 = bi;
        b = -1e30f; bi = 0;
        for (int r = 0; r < RGN; ++r) { if (r == c0 || r == c1) continue; if (mu_s[r] > b) { b = mu_s[r]; bi = r; } }
        cand[2] = bi; muk[2] = b;
    }
    __syncthreads();
    float f2 = 0, d0 = 0, d1 = 0, d2 = 0, g0 = 0, g1 = 0, g2 = 0;
    const float* fp  = F + (size_t)i * DMODEL;
    const float* gp0 = G + (size_t)cand[0] * DMODEL;
    const float* gp1 = G + (size_t)cand[1] * DMODEL;
    const float* gp2 = G + (size_t)cand[2] * DMODEL;
    for (int d = t; d < DMODEL; d += 64) {
        float f = fp[d];
        f2 += f * f;
        float a = gp0[d]; d0 += f * a; g0 += a * a;
        float bb = gp1[d]; d1 += f * bb; g1 += bb * bb;
        float c = gp2[d]; d2 += f * c; g2 += c * c;
    }
    p_f2[t] = f2;
    p_dot[0][t] = d0; p_dot[1][t] = d1; p_dot[2][t] = d2;
    p_g2[0][t] = g0;  p_g2[1][t] = g1;  p_g2[2][t] = g2;
    __syncthreads();
    if (t == 0) {
        float F2 = 0, D0 = 0, D1 = 0, D2 = 0, G0 = 0, G1 = 0, G2 = 0;
        for (int u = 0; u < 64; ++u) {
            F2 += p_f2[u];
            D0 += p_dot[0][u]; D1 += p_dot[1][u]; D2 += p_dot[2][u];
            G0 += p_g2[0][u];  G1 += p_g2[1][u];  G2 += p_g2[2][u];
        }
        float fn = fmaxf(sqrtf(F2), 1e-8f);
        float s0 = D0 / (fn * fmaxf(sqrtf(G0), 1e-8f));
        float s1 = D1 / (fn * fmaxf(sqrtf(G1), 1e-8f));
        float s2 = D2 / (fn * fmaxf(sqrtf(G2), 1e-8f));
        float m = fmaxf(s0, fmaxf(s1, s2));
        float e0 = __expf(s0 - m), e1 = __expf(s1 - m), e2 = __expf(s2 - m);
        float inv = 1.0f / (e0 + e1 + e2);
        float w0 = e0 * inv * muk[0], w1 = e1 * inv * muk[1], w2 = e2 * inv * muk[2];
        int best = 0; float wb = w0;
        if (w1 > wb) { wb = w1; best = 1; }
        if (w2 > wb) { wb = w2; best = 2; }
        int r = cand[best];
        assigned[i] = r;
        atomicAdd(&ctrl->regCount[r], 1);
    }
}

__global__ void k_compact(Ctrl* __restrict__ ctrl) {
    if (blockIdx.x == 0 && threadIdx.x == 0) {
        for (int s = 0; s < VMAX; ++s) { ctrl->slotCount[s] = 0; ctrl->cursor[s] = 0; }
        int V = 0;
        for (int r = 0; r < RGN; ++r) {
            if (ctrl->regCount[r] > 0) {
                ctrl->region2slot[r] = V;
                int c = ctrl->regCount[r];
                if (c > SMAX - 1) c = SMAX - 1;
                ctrl->slotCount[V] = c;
                V++;
            } else {
                ctrl->region2slot[r] = -1;
            }
        }
        ctrl->V = V;
    }
}

__global__ void k_scatter(const float* __restrict__ F, const int* __restrict__ assigned,
                          Ctrl* __restrict__ ctrl, int* __restrict__ slotpt,
                          float* __restrict__ X) {
    __shared__ int s_slot, s_pos;
    int i = blockIdx.x;
    if (threadIdx.x == 0) {
        int r = assigned[i];
        int slot = ctrl->region2slot[r];
        s_slot = slot;
        slotpt[i] = slot;
        s_pos = atomicAdd(&ctrl->cursor[slot], 1);
    }
    __syncthreads();
    if (s_pos < SMAX - 1) {
        float* dst = X + ((size_t)s_slot * SMAX + 1 + s_pos) * DMODEL;
        const float* src = F + (size_t)i * DMODEL;
        for (int d = threadIdx.x; d < DMODEL; d += 256) dst[d] = src[d];
    }
}

__global__ void k_cls(const float* __restrict__ cls, float* __restrict__ X) {
    int s = blockIdx.x;
    for (int d = threadIdx.x; d < DMODEL; d += 256)
        X[(size_t)s * SMAX * DMODEL + d] = cls[d];
}

// ---------------- encoder GEMM (async global->LDS stage + bf16 WMMA) ----------------
// Block = 256 threads = 8 waves (4x2). Block tile 128x128, wave tile 32x64.
// Per k-step: 128x32 A tile and 128x32 B tile staged via GLOBAL_LOAD_ASYNC_TO_LDS_B128
// (512 x 16B lane-transfers each), s_wait_asynccnt 0 + barrier, then all WMMA
// fragments are LDS-resident (ds_load_b128).
template <int OUT_BF16, int ACT_GELU>
__global__ void k_gemm(const bf16* __restrict__ A, int lda,
                       const bf16* __restrict__ B, int ldb,
                       const float* __restrict__ bias,
                       void* __restrict__ Cout, int ldc, int K) {
    extern __shared__ char gsm[];
    bf16* sA = (bf16*)gsm;                      // [128][32]
    bf16* sB = (bf16*)(gsm + 128 * 32 * 2);     // [128][32]
    unsigned tid = threadIdx.x, lane = tid & 31, wv = tid >> 5;
    int mBase = blockIdx.x * 128;
    int nBase = blockIdx.y * 128;
    int mW = (int)(wv >> 1) * 32;
    int nW = (int)(wv & 1) * 64;
    unsigned sAoff = lds_byte_off(sA);
    unsigned sBoff = lds_byte_off(sB);
    v8f acc[2][4] = {};
    for (int k0 = 0; k0 < K; k0 += 32) {
        __builtin_prefetch(A + (size_t)(mBase + (int)(tid >> 1)) * lda + k0 + 32, 0, 1);
#pragma unroll
        for (int it = 0; it < 2; ++it) {
            unsigned c = tid + it * 256u;       // 0..511 : 16B chunk id
            unsigned row = c >> 2, sub = c & 3; // 4 chunks per 64B tile row
            const bf16* ga = A + (size_t)(mBase + (int)row) * lda + k0 + sub * 8;
            const bf16* gb = B + (size_t)(nBase + (int)row) * ldb + k0 + sub * 8;
            async_g2l_b128(sAoff + c * 16u, ga);
            async_g2l_b128(sBoff + c * 16u, gb);
        }
        wait_async0();
        __syncthreads();
        v16bf a0 = frag_a_rowmajor(sA, 32, mW, 0, lane);
        v16bf a1 = frag_a_rowmajor(sA, 32, mW + 16, 0, lane);
#pragma unroll
        for (int j = 0; j < 4; ++j) {
            v16bf b = frag_b_nk(sB, 32, nW + j * 16, 0, lane);
            acc[0][j] = wmma_bf(a0, b, acc[0][j]);
            acc[1][j] = wmma_bf(a1, b, acc[1][j]);
        }
        __syncthreads();
    }
#pragma unroll
    for (int i = 0; i < 2; ++i)
#pragma unroll
        for (int j = 0; j < 4; ++j)
#pragma unroll
            for (int v = 0; v < 8; ++v) {
                int row = mBase + mW + i * 16 + v + (int)((lane >> 4) << 3);
                int col = nBase + nW + j * 16 + (int)(lane & 15);
                float x = acc[i][j][v] + bias[col];
                if (ACT_GELU) x = 0.5f * x * (1.0f + erff(x * 0.70710678f));
                if (OUT_BF16) ((bf16*)Cout)[(size_t)row * ldc + col] = (bf16)x;
                else          ((float*)Cout)[(size_t)row * ldc + col] = x;
            }
}

// ---------------- attention: one wave per (region, head, 16-query tile) ----------------
// scores (WMMA) -> LDS, masked softmax (wave32 shfl), probs as bf16 in LDS, attnxV (WMMA).
__global__ void k_attention(const bf16* __restrict__ qkv, const Ctrl* __restrict__ ctrl,
                            float* __restrict__ O) {
    extern __shared__ char smem[];
    float* sc = (float*)smem;                                     // [16][SMAX] scores
    bf16* pb = (bf16*)(smem + (size_t)16 * SMAX * sizeof(float)); // [16][SMAX] probs
    int v = blockIdx.x, h = blockIdx.y;
    unsigned lane = threadIdx.x & 31;
    int q0 = blockIdx.z * 16;
    int cnt = ctrl->slotCount[v]; // key j valid iff j <= cnt (j==0 is CLS)
    const bf16* qp = qkv + (size_t)v * SMAX * QKVLD + h * HDIM;
    const bf16* kp = qp + DMODEL;
    const bf16* vp = qp + 2 * DMODEL;
    const float scale = 0.10206207f; // 1/sqrt(96)

    v16bf aq[3];
#pragma unroll
    for (int d = 0; d < 3; ++d) aq[d] = frag_a_rowmajor(qp, QKVLD, q0, d * 32, lane);

    for (int kb = 0; kb < SMAX / 16; ++kb) {
        v8f c = {};
#pragma unroll
        for (int d = 0; d < 3; ++d) {
            v16bf b = frag_b_nk(kp, QKVLD, kb * 16, d * 32, lane);
            c = wmma_bf(aq[d], b, c);
        }
        int col = kb * 16 + (int)(lane & 15);
        bool masked = col > cnt;
#pragma unroll
        for (int r8 = 0; r8 < 8; ++r8) {
            int row = r8 + (int)((lane >> 4) << 3);
            sc[row * SMAX + col] = masked ? -1e9f : c[r8] * scale;
        }
    }
    __syncthreads();
    {
        int row = (int)(lane >> 1), half = (int)(lane & 1);
        const float* rs = sc + row * SMAX;
        float m = -1e30f;
        for (int j = half * (SMAX / 2); j < (half + 1) * (SMAX / 2); ++j) m = fmaxf(m, rs[j]);
        m = fmaxf(m, __shfl_xor(m, 1, 32));
        float s = 0.f;
        for (int j = half * (SMAX / 2); j < (half + 1) * (SMAX / 2); ++j) s += __expf(rs[j] - m);
        s += __shfl_xor(s, 1, 32);
        float inv = 1.0f / s;
        bf16* rp = pb + row * SMAX;
        for (int j = half * (SMAX / 2); j < (half + 1) * (SMAX / 2); ++j)
            rp[j] = (bf16)(__expf(rs[j] - m) * inv);
    }
    __syncthreads();
    v8f oacc[6] = {};
    for (int kc = 0; kc < SMAX; kc += 32) {
        v16bf ap = frag_a_rowmajor(pb, SMAX, 0, kc, lane); // LDS A-fragment (ds_load)
#pragma unroll
        for (int j = 0; j < 6; ++j) {
            v16bf b = frag_b_kn(vp, QKVLD, kc, j * 16, lane);
            oacc[j] = wmma_bf(ap, b, oacc[j]);
        }
    }
#pragma unroll
    for (int j = 0; j < 6; ++j)
#pragma unroll
        for (int r8 = 0; r8 < 8; ++r8) {
            int row = q0 + r8 + (int)((lane >> 4) << 3);
            int col = j * 16 + (int)(lane & 15);
            O[((size_t)v * SMAX + row) * DMODEL + h * HDIM + col] = oacc[j][r8];
        }
}

// ---------------- residual + LayerNorm ----------------
__global__ void k_addln(const float* __restrict__ A, const float* __restrict__ B,
                        const float* __restrict__ w, const float* __restrict__ b,
                        float* __restrict__ out) {
    __shared__ float s1[256], s2[256];
    int row = blockIdx.x;
    const float* pa = A + (size_t)row * DMODEL;
    const float* pr = B + (size_t)row * DMODEL;
    float a1 = 0, a2 = 0;
    for (int d = threadIdx.x; d < DMODEL; d += 256) {
        float x = pa[d] + pr[d];
        a1 += x; a2 += x * x;
    }
    s1[threadIdx.x] = a1; s2[threadIdx.x] = a2;
    __syncthreads();
    for (int off = 128; off > 0; off >>= 1) {
        if (threadIdx.x < off) { s1[threadIdx.x] += s1[threadIdx.x + off]; s2[threadIdx.x] += s2[threadIdx.x + off]; }
        __syncthreads();
    }
    float mean = s1[0] / DMODEL;
    float var = s2[0] / DMODEL - mean * mean;
    float inv = rsqrtf(var + LNEPS);
    for (int d = threadIdx.x; d < DMODEL; d += 256) {
        float x = pa[d] + pr[d];
        out[(size_t)row * DMODEL + d] = (x - mean) * inv * w[d] + b[d];
    }
}

__global__ void k_copyout(const float* __restrict__ FIN, const int* __restrict__ slotpt,
                          const Ctrl* __restrict__ ctrl, float* __restrict__ out) {
    int V = ctrl->V;
    int b = blockIdx.x;
    if (b < 64) {
        if (b < V)
            for (int d = threadIdx.x; d < DMODEL; d += 256)
                out[(size_t)b * DMODEL + d] = FIN[(size_t)b * SMAX * DMODEL + d];
    } else {
        int i = (b - 64) * 256 + threadIdx.x;
        if (i < NPATCH) out[(size_t)V * DMODEL + i] = (float)slotpt[i];
    }
}

// ---------------- host orchestration ----------------
extern "C" void kernel_launch(void* const* d_in, const int* in_sizes, int n_in,
                              void* d_out, int out_size, void* d_ws, size_t ws_size,
                              hipStream_t stream) {
    (void)in_sizes; (void)n_in; (void)out_size; (void)ws_size;
    const float* F    = (const float*)d_in[0];
    const float* T    = (const float*)d_in[1];
    const float* cls  = (const float*)d_in[2];
    const float* Wqkv = (const float*)d_in[3];
    const float* bqkv = (const float*)d_in[4];
    const float* Wout = (const float*)d_in[5];
    const float* bout = (const float*)d_in[6];
    const float* Wl1  = (const float*)d_in[7];
    const float* bl1  = (const float*)d_in[8];
    const float* Wl2  = (const float*)d_in[9];
    const float* bl2  = (const float*)d_in[10];
    const float* ln1w = (const float*)d_in[11];
    const float* ln1b = (const float*)d_in[12];
    const float* ln2w = (const float*)d_in[13];
    const float* ln2b = (const float*)d_in[14];
    float* out = (float*)d_out;

    char* ws = (char*)d_ws;
    Ctrl* ctrl    = (Ctrl*)ws;
    int* assigned = (int*)(ws + OFF_ASSIGN);
    int* slotpt   = (int*)(ws + OFF_SLOTPT);
    float* G      = (float*)(ws + OFF_G);
    float* X      = (float*)(ws + OFF_X);
    float* H      = (float*)(ws + OFF_H);
    float* Obuf   = (float*)(ws + OFF_O);
    bf16* qkvbf   = (bf16*)(ws + OFF_QKV);
    float* TMP    = (float*)(ws + OFF_QKV);   // reuses qkv region after attention
    bf16* Abf     = (bf16*)(ws + OFF_ABF);
    bf16* FF1     = (bf16*)(ws + OFF_FF1);
    bf16* WqkvB   = (bf16*)(ws + OFF_WBF) + WOFF_QKV;
    bf16* WoutB   = (bf16*)(ws + OFF_WBF) + WOFF_OUT;
    bf16* Wl1B    = (bf16*)(ws + OFF_WBF) + WOFF_L1;
    bf16* Wl2B    = (bf16*)(ws + OFF_WBF) + WOFF_L2;

    // init: control block + token tensor zeroed every call (deterministic)
    k_zero_u32<<<256, 256, 0, stream>>>((unsigned*)ws, (long)((1 << 20) / 4));
    k_zero_u32<<<2048, 256, 0, stream>>>((unsigned*)X, (long)ROWS * DMODEL);

    // routing
    k_minmax<<<64, 256, 0, stream>>>(T, ctrl);
    k_finalize<<<1, 64, 0, stream>>>(ctrl);
    k_accumG<<<NPATCH, 256, 0, stream>>>(F, T, ctrl, G);
    k_normG<<<RGN, 256, 0, stream>>>(ctrl, G);
    k_maxdist<<<NPATCH / 256, 256, 0, stream>>>(T, ctrl);
    k_assign<<<NPATCH, 64, 0, stream>>>(F, T, ctrl, G, assigned);
    k_compact<<<1, 1, 0, stream>>>(ctrl);
    k_scatter<<<NPATCH, 256, 0, stream>>>(F, assigned, ctrl, slotpt, X);
    k_cls<<<VMAX, 256, 0, stream>>>(cls, X);

    // weight conversion to bf16
    k_f32_to_bf16<<<1024, 256, 0, stream>>>(Wqkv, WqkvB, (long)QKVLD * DMODEL);
    k_f32_to_bf16<<<512, 256, 0, stream>>>(Wout, WoutB, (long)DMODEL * DMODEL);
    k_f32_to_bf16<<<1024, 256, 0, stream>>>(Wl1, Wl1B, (long)DFF * DMODEL);
    k_f32_to_bf16<<<1024, 256, 0, stream>>>(Wl2, Wl2B, (long)DMODEL * DFF);

    const size_t gemm_lds = (size_t)2 * 128 * 32 * sizeof(bf16);  // 16 KB

    // QKV projection
    k_f32_to_bf16<<<2048, 256, 0, stream>>>(X, Abf, (long)ROWS * DMODEL);
    {
        dim3 g(ROWS / 128, QKVLD / 128);
        k_gemm<1, 0><<<g, 256, gemm_lds, stream>>>(Abf, DMODEL, WqkvB, DMODEL, bqkv,
                                                   (void*)qkvbf, QKVLD, DMODEL);
    }
    // attention (one wave per region/head/16-query tile)
    {
        dim3 g(VMAX, NHEAD, SMAX / 16);
        size_t smem = (size_t)16 * SMAX * sizeof(float) + (size_t)16 * SMAX * sizeof(bf16);
        k_attention<<<g, 32, smem, stream>>>(qkvbf, ctrl, Obuf);
    }
    // out projection + residual + LN1
    k_f32_to_bf16<<<2048, 256, 0, stream>>>(Obuf, Abf, (long)ROWS * DMODEL);
    {
        dim3 g(ROWS / 128, DMODEL / 128);
        k_gemm<0, 0><<<g, 256, gemm_lds, stream>>>(Abf, DMODEL, WoutB, DMODEL, bout,
                                                   (void*)TMP, DMODEL, DMODEL);
    }
    k_addln<<<ROWS, 256, 0, stream>>>(X, TMP, ln1w, ln1b, H);

    // FFN
    k_f32_to_bf16<<<2048, 256, 0, stream>>>(H, Abf, (long)ROWS * DMODEL);
    {
        dim3 g(ROWS / 128, DFF / 128);
        k_gemm<1, 1><<<g, 256, gemm_lds, stream>>>(Abf, DMODEL, Wl1B, DMODEL, bl1,
                                                   (void*)FF1, DFF, DMODEL);
    }
    {
        dim3 g(ROWS / 128, DMODEL / 128);
        k_gemm<0, 0><<<g, 256, gemm_lds, stream>>>(FF1, DFF, Wl2B, DFF, bl2,
                                                   (void*)TMP, DMODEL, DFF);
    }
    k_addln<<<ROWS, 256, 0, stream>>>(H, TMP, ln2w, ln2b, Obuf);

    // outputs: CLS rows + region indices
    k_copyout<<<128, 256, 0, stream>>>(Obuf, slotpt, ctrl, out);
}
